// MMDoubleStreamBlock_88914412962460
// MI455X (gfx1250) — compile-verified, matrix operand order checked by reading, not run on hardware
//
#include <hip/hip_runtime.h>
#include <hip/hip_bf16.h>
#include <math.h>

// ---------------- problem constants ----------------
#define S_LEN   2048
#define T_LEN   256
#define D_MODEL 3072
#define NHEADS  24
#define HEADD   128
#define FF_DIM  12288
#define KV_LEN  (S_LEN + T_LEN)   // 2304

typedef __bf16 bf16;
typedef __attribute__((ext_vector_type(16))) __bf16 v16bf;
typedef __attribute__((ext_vector_type(8)))  float  v8f;

union U16 { v16bf v; unsigned u[8]; };

static __device__ __forceinline__ unsigned pack_bf16(float a, float b) {
    union { __bf16 h[2]; unsigned u; } x;
    x.h[0] = (__bf16)a; x.h[1] = (__bf16)b;
    return x.u;
}

#define WMMA_BF16(A, B, C) __builtin_amdgcn_wmma_f32_16x16x32_bf16(false, (A), false, (B), (short)0, (C), false, false)

// ---------------- 1) silu(vec) ----------------
__global__ void k_silu(const float* __restrict__ vec, float* __restrict__ sv) {
    int i = blockIdx.x * 256 + threadIdx.x;
    float v = vec[i];
    sv[i] = v / (1.0f + __expf(-v));
}

// ---------------- 2) mod = silu(vec) @ mod_w + mod_b (GEMV) ----------------
__global__ void k_mod(const float* __restrict__ sv, const float* __restrict__ mod_w,
                      const float* __restrict__ mod_b, float* __restrict__ mod) {
    __shared__ float svs[D_MODEL];
    int t = threadIdx.x;
    #pragma unroll
    for (int i = 0; i < 12; ++i) svs[t + i * 256] = sv[t + i * 256];
    __syncthreads();
    int j = blockIdx.x * 256 + t;
    float acc = mod_b[j];
    for (int i = 0; i < D_MODEL; ++i)
        acc += svs[i] * mod_w[(size_t)i * (6 * D_MODEL) + j];
    mod[j] = acc;
}

// ---------------- 3) weight transpose + f32->bf16:  Wt[n][k] = W[k][n] ----------
__global__ void k_wt(const float* __restrict__ W, bf16* __restrict__ Wt, int K, int N) {
    __shared__ float tile[32][33];
    int n0 = blockIdx.x * 32, k0 = blockIdx.y * 32;
    int tc = threadIdx.x & 31, tr = threadIdx.x >> 5;   // 8 rows of 32
    #pragma unroll
    for (int i = 0; i < 4; ++i) {
        int r = tr + i * 8;
        tile[r][tc] = W[(size_t)(k0 + r) * N + n0 + tc];
    }
    __syncthreads();
    #pragma unroll
    for (int i = 0; i < 4; ++i) {
        int r = tr + i * 8;     // local n row
        Wt[(size_t)(n0 + r) * K + k0 + tc] = (bf16)tile[tc][r];
    }
}

// ---------------- 4) x1 = LN(img)*(1+a_scale)+a_shift  -> bf16 ----------------
__global__ void k_ln_mod(const float* __restrict__ img, const float* __restrict__ mod,
                         bf16* __restrict__ x1) {
    int row = blockIdx.x, t = threadIdx.x;
    __shared__ float r1[256], r2[256];
    float vals[12]; float s = 0.f, s2 = 0.f;
    const float* ir = img + (size_t)row * D_MODEL;
    #pragma unroll
    for (int i = 0; i < 12; ++i) { float v = ir[t + i * 256]; vals[i] = v; s += v; s2 += v * v; }
    r1[t] = s; r2[t] = s2; __syncthreads();
    for (int st = 128; st > 0; st >>= 1) {
        if (t < st) { r1[t] += r1[t + st]; r2[t] += r2[t + st]; }
        __syncthreads();
    }
    float mean = r1[0] * (1.0f / D_MODEL);
    float var  = r2[0] * (1.0f / D_MODEL) - mean * mean;
    float rstd = rsqrtf(var + 1e-6f);
    #pragma unroll
    for (int i = 0; i < 12; ++i) {
        int c = t + i * 256;
        float xh = (vals[i] - mean) * rstd;
        x1[(size_t)row * D_MODEL + c] = (bf16)(xh * (1.0f + mod[D_MODEL + c]) + mod[c]);
    }
}

// ---------------- 5) GEMM: C = A_bf16[MxK] @ Wt_bf16[N][K]^T + bias -------------
// MODE 0: f32 out.  MODE 1: bf16 out = gelu_tanh.
// Block 256 thr = 8 waves, tile 128(M) x 128(N), BK=32, double-buffered LDS
// filled with GLOBAL_LOAD_ASYNC_TO_LDS_B64 (ASYNCcnt-tracked).
// Wave (wm = wid>>1, wn = wid&1) owns 32(M) x 64(N) = 2x4 WMMA tiles.
#define LDSTR 18    // uints per LDS row (72B: 8B-aligned rows, conflict-free frag reads)
template <int MODE>
__global__ void k_gemm(const bf16* __restrict__ A, const bf16* __restrict__ Wt,
                       const float* __restrict__ bias, void* __restrict__ Cout,
                       int M, int N, int K) {
    __shared__ unsigned As_u[2][128 * LDSTR];
    __shared__ unsigned Bs_u[2][128 * LDSTR];
    const int t = threadIdx.x;
    const int lane = t & 31, wid = t >> 5;
    const int waveM = wid >> 1, waveN = wid & 1;
    const int m0 = blockIdx.y * 128;
    const int n0 = blockIdx.x * 128;

    v8f c[2][4] = {};

    // async-issue one 128x32 A tile + 128x32 B tile into LDS buffer b.
    // Each thread issues 4+4 b64 async loads (8B per lane per op).
    auto issue_tile = [&](int k0, int b) {
        #pragma unroll
        for (int i = 0; i < 4; ++i) {
            int id = t + i * 256;          // 0..1023
            int r = id >> 3, cc = id & 7;  // row 0..127, b64-chunk 0..7
            unsigned lA = (unsigned)(size_t)&As_u[b][r * LDSTR + 2 * cc];
            unsigned gA = (unsigned)(((((size_t)(m0 + r)) * K + k0) >> 1) + 2 * cc) * 4u;
            asm volatile("global_load_async_to_lds_b64 %0, %1, %2"
                         :: "v"(lA), "v"(gA), "s"(A) : "memory");
            unsigned lB = (unsigned)(size_t)&Bs_u[b][r * LDSTR + 2 * cc];
            unsigned gB = (unsigned)(((((size_t)(n0 + r)) * K + k0) >> 1) + 2 * cc) * 4u;
            asm volatile("global_load_async_to_lds_b64 %0, %1, %2"
                         :: "v"(lB), "v"(gB), "s"(Wt) : "memory");
        }
    };

    const int NT = K >> 5;
    issue_tile(0, 0);

    for (int kt = 0; kt < NT; ++kt) {
        const int cur = kt & 1;
        asm volatile("s_wait_asynccnt 0x0" ::: "memory");
        __syncthreads();
        if (kt + 1 < NT) issue_tile((kt + 1) << 5, cur ^ 1);

        // fragments: lanes 0-15 hold K 0-7 (u0..3) & 16-23 (u4..7); lanes 16-31: K 8-15 & 24-31
        U16 a[2], b[4];
        const int kb2 = (lane >> 4) * 4;
        #pragma unroll
        for (int mt = 0; mt < 2; ++mt) {
            int mr = (waveM * 32 + mt * 16 + (lane & 15)) * LDSTR;
            #pragma unroll
            for (int v = 0; v < 4; ++v) {
                a[mt].u[v] = As_u[cur][mr + kb2 + v]; a[mt].u[v + 4] = As_u[cur][mr + 8 + kb2 + v];
            }
        }
        #pragma unroll
        for (int nt = 0; nt < 4; ++nt) {
            int nr = (waveN * 64 + nt * 16 + (lane & 15)) * LDSTR;
            #pragma unroll
            for (int v = 0; v < 4; ++v) {
                b[nt].u[v] = Bs_u[cur][nr + kb2 + v]; b[nt].u[v + 4] = Bs_u[cur][nr + 8 + kb2 + v];
            }
        }

        #pragma unroll
        for (int mt = 0; mt < 2; ++mt)
        #pragma unroll
        for (int nt = 0; nt < 4; ++nt)
            c[mt][nt] = WMMA_BF16(a[mt].v, b[nt].v, c[mt][nt]);
    }

    // epilogue: C/D layout: lanes 0-15 -> M=v, lanes 16-31 -> M=v+8; N = lane%16
    const int ml = (lane < 16) ? 0 : 8;
    const int nl = lane & 15;
    #pragma unroll
    for (int mt = 0; mt < 2; ++mt)
    #pragma unroll
    for (int nt = 0; nt < 4; ++nt) {
        v8f cc = c[mt][nt];
        int col = n0 + waveN * 64 + nt * 16 + nl;
        float bs = bias[col];
        #pragma unroll
        for (int v = 0; v < 8; ++v) {
            int row = m0 + waveM * 32 + mt * 16 + ml + v;
            float val = cc[v] + bs;
            if (MODE == 0) {
                ((float*)Cout)[(size_t)row * N + col] = val;
            } else {
                float g = 0.5f * val * (1.0f + tanhf(0.7978845608f * (val + 0.044715f * val * val * val)));
                ((bf16*)Cout)[(size_t)row * N + col] = (bf16)g;
            }
        }
    }
}

// ---------------- 6) RMSNorm(q,k) + interleaved RoPE, in place on qkv f32 --------
__global__ void k_qknorm_rope(float* __restrict__ qkv, const float* __restrict__ qn,
                              const float* __restrict__ kn, const float* __restrict__ cosb,
                              const float* __restrict__ sinb) {
    int wid = threadIdx.x >> 5, lane = threadIdx.x & 31;
    int idx = blockIdx.x * 8 + wid;
    int s = idx / NHEADS, h = idx % NHEADS;
    #pragma unroll
    for (int which = 0; which < 2; ++which) {
        float* base = qkv + (size_t)s * (3 * D_MODEL) + which * D_MODEL + h * HEADD;
        const float* w = which ? kn : qn;
        float4 x = *(const float4*)(base + 4 * lane);
        float ss = x.x * x.x + x.y * x.y + x.z * x.z + x.w * x.w;
        #pragma unroll
        for (int m = 16; m >= 1; m >>= 1) ss += __shfl_xor(ss, m, 32);
        float r = rsqrtf(ss * (1.0f / HEADD) + 1e-6f);
        float a  = x.x * r * w[4 * lane + 0];
        float b  = x.y * r * w[4 * lane + 1];
        float c2 = x.z * r * w[4 * lane + 2];
        float d2 = x.w * r * w[4 * lane + 3];
        float c0 = cosb[s * 64 + 2 * lane],     s0 = sinb[s * 64 + 2 * lane];
        float c1 = cosb[s * 64 + 2 * lane + 1], s1 = sinb[s * 64 + 2 * lane + 1];
        float4 o;
        o.x = a  * c0 - b  * s0;  o.y = b  * c0 + a  * s0;
        o.z = c2 * c1 - d2 * s1;  o.w = d2 * c1 + c2 * s1;
        *(float4*)(base + 4 * lane) = o;
    }
}

// ---------------- 7) WMMA flash attention: block = (head, 64 queries) -----------
__global__ void k_attn(const float* __restrict__ qkv, const float* __restrict__ k_txt,
                       const float* __restrict__ v_txt, bf16* __restrict__ attn_out) {
    const int h  = blockIdx.x;      // 0..23
    const int qb = blockIdx.y;      // 0..31 (64 queries each)
    const int t = threadIdx.x;
    const int lane = t & 31, wid = t >> 5;

    __shared__ unsigned Qs_u[64 * 65];    // [q][64 d-pairs] (+1 pad)
    __shared__ unsigned Ks_u[64 * 65];    // [kv][64 d-pairs]
    __shared__ unsigned Vt_u[128 * 33];   // [d][32 kv-pairs] (+1 pad)
    __shared__ unsigned Pb_u[64 * 33];    // [q][32 kv-pairs]
    __shared__ float Sc[64 * 65];         // scores f32
    __shared__ float Mrow[64], Lrow[64], Arow[64];
    bf16* Vt_h = (bf16*)Vt_u;
    bf16* Pb_h = (bf16*)Pb_u;

    // load Q tile (64 x 128) as bf16 pairs
    #pragma unroll
    for (int i = 0; i < 16; ++i) {
        int idx = t + i * 256;          // 0..4095
        int q = idx >> 6, u = idx & 63;
        const float2 f = *(const float2*)(qkv + (size_t)(qb * 64 + q) * (3 * D_MODEL) + h * HEADD + 2 * u);
        Qs_u[q * 65 + u] = pack_bf16(f.x, f.y);
    }
    if (t < 64) { Mrow[t] = -3.0e38f; Lrow[t] = 0.f; }

    v8f o[2][2] = {};
    const float scale = 0.08838834764831845f;   // 1/sqrt(128)
    const int kb2 = (lane >> 4) * 4;
    const int nl = lane & 15;
    const int ml = (lane < 16) ? 0 : 8;

    for (int kb = 0; kb < KV_LEN / 64; ++kb) {
        __syncthreads();
        // stage K chunk as pairs; stage V chunk transposed (b16 scatter)
        #pragma unroll
        for (int i = 0; i < 16; ++i) {
            int idx = t + i * 256;
            int r = idx >> 6, u = idx & 63;
            int j = kb * 64 + r;
            float2 f;
            if (j < S_LEN) f = *(const float2*)(qkv + (size_t)j * (3 * D_MODEL) + D_MODEL + h * HEADD + 2 * u);
            else           f = *(const float2*)(k_txt + ((size_t)(j - S_LEN) * NHEADS + h) * HEADD + 2 * u);
            Ks_u[r * 65 + u] = pack_bf16(f.x, f.y);
        }
        #pragma unroll
        for (int i = 0; i < 32; ++i) {
            int idx = t + i * 256;          // 0..8191
            int kv = idx >> 7, d = idx & 127;
            int j = kb * 64 + kv;
            float v;
            if (j < S_LEN) v = qkv[(size_t)j * (3 * D_MODEL) + 2 * D_MODEL + h * HEADD + d];
            else           v = v_txt[((size_t)(j - S_LEN) * NHEADS + h) * HEADD + d];
            Vt_h[d * 66 + kv] = (bf16)v;
        }
        __syncthreads();

        // ---- QK^T: wave = (wid>>2)*32 q rows x (wid&3)*16 kv cols ----
        {
            const int qsub = (wid >> 2) * 32;
            const int ksub = (wid & 3) * 16;
            v8f s0 = {}, s1 = {};
            #pragma unroll
            for (int kk = 0; kk < 4; ++kk) {      // head-dim steps of 32
                U16 a0, a1, bb;
                int pbase = kk * 16;
                int mr0 = (qsub + nl) * 65;
                int mr1 = (qsub + 16 + nl) * 65;
                int nr  = (ksub + nl) * 65;
                #pragma unroll
                for (int v = 0; v < 4; ++v) {
                    a0.u[v] = Qs_u[mr0 + pbase + kb2 + v]; a0.u[v + 4] = Qs_u[mr0 + pbase + 8 + kb2 + v];
                    a1.u[v] = Qs_u[mr1 + pbase + kb2 + v]; a1.u[v + 4] = Qs_u[mr1 + pbase + 8 + kb2 + v];
                    bb.u[v] = Ks_u[nr  + pbase + kb2 + v]; bb.u[v + 4] = Ks_u[nr  + pbase + 8 + kb2 + v];
                }
                s0 = WMMA_BF16(a0.v, bb.v, s0);
                s1 = WMMA_BF16(a1.v, bb.v, s1);
            }
            #pragma unroll
            for (int v = 0; v < 8; ++v) {
                Sc[(qsub + ml + v) * 65 + ksub + nl]      = s0[v] * scale;
                Sc[(qsub + 16 + ml + v) * 65 + ksub + nl] = s1[v] * scale;
            }
        }
        __syncthreads();

        // ---- online softmax over this 64-wide chunk (one thread per q row) ----
        if (t < 64) {
            float m = Mrow[t], mx = m;
            #pragma unroll
            for (int c2 = 0; c2 < 64; ++c2) mx = fmaxf(mx, Sc[t * 65 + c2]);
            float alpha = __expf(m - mx);
            float sum = 0.f;
            #pragma unroll
            for (int c2 = 0; c2 < 64; ++c2) {
                float p = __expf(Sc[t * 65 + c2] - mx);
                Pb_h[t * 66 + c2] = (bf16)p;
                sum += p;
            }
            Lrow[t] = Lrow[t] * alpha + sum; Mrow[t] = mx; Arow[t] = alpha;
        }
        __syncthreads();

        // ---- rescale accumulators, then PV: wave = (wid>>2)*32 q x (wid&3)*32 d ----
        {
            const int qsub = (wid >> 2) * 32;
            const int dsub = (wid & 3) * 32;
            #pragma unroll
            for (int mt = 0; mt < 2; ++mt) {
                #pragma unroll
                for (int v = 0; v < 8; ++v) {
                    float al = Arow[qsub + mt * 16 + ml + v];
                    o[mt][0][v] *= al;
                    o[mt][1][v] *= al;
                }
            }
            #pragma unroll
            for (int kk = 0; kk < 2; ++kk) {      // kv steps of 32
                U16 a0, a1, b0, b1;
                int pbase = kk * 16;
                int mr0 = (qsub + nl) * 33;
                int mr1 = (qsub + 16 + nl) * 33;
                int nr0 = (dsub + nl) * 33;
                int nr1 = (dsub + 16 + nl) * 33;
                #pragma unroll
                for (int v = 0; v < 4; ++v) {
                    a0.u[v] = Pb_u[mr0 + pbase + kb2 + v]; a0.u[v + 4] = Pb_u[mr0 + pbase + 8 + kb2 + v];
                    a1.u[v] = Pb_u[mr1 + pbase + kb2 + v]; a1.u[v + 4] = Pb_u[mr1 + pbase + 8 + kb2 + v];
                    b0.u[v] = Vt_u[nr0 + pbase + kb2 + v]; b0.u[v + 4] = Vt_u[nr0 + pbase + 8 + kb2 + v];
                    b1.u[v] = Vt_u[nr1 + pbase + kb2 + v]; b1.u[v + 4] = Vt_u[nr1 + pbase + 8 + kb2 + v];
                }
                o[0][0] = WMMA_BF16(a0.v, b0.v, o[0][0]);
                o[0][1] = WMMA_BF16(a0.v, b1.v, o[0][1]);
                o[1][0] = WMMA_BF16(a1.v, b0.v, o[1][0]);
                o[1][1] = WMMA_BF16(a1.v, b1.v, o[1][1]);
            }
        }
    }
    __syncthreads();

    // ---- epilogue: out = o / l ----
    {
        const int qsub = (wid >> 2) * 32;
        const int dsub = (wid & 3) * 32;
        #pragma unroll
        for (int mt = 0; mt < 2; ++mt)
        #pragma unroll
        for (int nt = 0; nt < 2; ++nt) {
            #pragma unroll
            for (int v = 0; v < 8; ++v) {
                int q = qsub + mt * 16 + ml + v;
                int d = dsub + nt * 16 + nl;
                attn_out[(size_t)(qb * 64 + q) * D_MODEL + h * HEADD + d] = (bf16)(o[mt][nt][v] / Lrow[q]);
            }
        }
    }
}

// ---------------- 8) resid = img + a_gate*proj ; x2 = LN(resid)*(1+m_scale)+m_shift ----
__global__ void k_resid_ln(const float* __restrict__ img, const float* __restrict__ proj,
                           const float* __restrict__ mod, float* __restrict__ resid,
                           bf16* __restrict__ x2) {
    int row = blockIdx.x, t = threadIdx.x;
    __shared__ float r1[256], r2[256];
    float vals[12]; float s = 0.f, s2 = 0.f;
    #pragma unroll
    for (int i = 0; i < 12; ++i) {
        int c = t + i * 256;
        float v = img[(size_t)row * D_MODEL + c] + mod[2 * D_MODEL + c] * proj[(size_t)row * D_MODEL + c];
        vals[i] = v; resid[(size_t)row * D_MODEL + c] = v; s += v; s2 += v * v;
    }
    r1[t] = s; r2[t] = s2; __syncthreads();
    for (int st = 128; st > 0; st >>= 1) {
        if (t < st) { r1[t] += r1[t + st]; r2[t] += r2[t + st]; }
        __syncthreads();
    }
    float mean = r1[0] * (1.0f / D_MODEL);
    float var  = r2[0] * (1.0f / D_MODEL) - mean * mean;
    float rstd = rsqrtf(var + 1e-6f);
    #pragma unroll
    for (int i = 0; i < 12; ++i) {
        int c = t + i * 256;
        float xh = (vals[i] - mean) * rstd;
        x2[(size_t)row * D_MODEL + c] = (bf16)(xh * (1.0f + mod[4 * D_MODEL + c]) + mod[3 * D_MODEL + c]);
    }
}

// ---------------- 9) out = resid + m_gate * mlp2 ----------------
__global__ void k_final(const float* __restrict__ resid, const float* __restrict__ mlp2,
                        const float* __restrict__ mod, float* __restrict__ out) {
    int i = blockIdx.x * 256 + threadIdx.x;
    int c = i % D_MODEL;
    out[i] = resid[i] + mod[5 * D_MODEL + c] * mlp2[i];
}

// ---------------- workspace layout (bytes) ----------------
#define OFF_SV     ((size_t)0)
#define OFF_MOD    ((size_t)16384)
#define OFF_WQKV   ((size_t)262144)                  // 3072*9216  bf16 = 56623104
#define OFF_WPROJ  (OFF_WQKV  + (size_t)56623104)    // 3072*3072  bf16 = 18874368
#define OFF_WMLP1  (OFF_WPROJ + (size_t)18874368)    // 3072*12288 bf16 = 75497472
#define OFF_WMLP2  (OFF_WMLP1 + (size_t)75497472)    // 12288*3072 bf16 = 75497472
#define OFF_X1     (OFF_WMLP2 + (size_t)75497472)    // 2048*3072 bf16 = 12582912
#define OFF_QKV    (OFF_X1    + (size_t)12582912)    // 2048*9216 f32  = 75497472
#define OFF_ATTN   (OFF_QKV   + (size_t)75497472)    // 2048*3072 bf16 = 12582912
#define OFF_PROJ   (OFF_ATTN  + (size_t)12582912)    // 2048*3072 f32  = 25165824
#define OFF_RESID  (OFF_PROJ  + (size_t)25165824)    // 2048*3072 f32  = 25165824
#define OFF_X2     (OFF_RESID + (size_t)25165824)    // 2048*3072 bf16 = 12582912
#define OFF_H      OFF_QKV                           // reuse (qkv dead after attn)
#define OFF_MLP2   OFF_PROJ                          // reuse (proj dead after resid_ln)

extern "C" void kernel_launch(void* const* d_in, const int* in_sizes, int n_in,
                              void* d_out, int out_size, void* d_ws, size_t ws_size,
                              hipStream_t stream) {
    const float* img    = (const float*)d_in[0];
    const float* vec    = (const float*)d_in[1];
    const float* cosb   = (const float*)d_in[2];
    const float* sinb   = (const float*)d_in[3];
    const float* k_txt  = (const float*)d_in[4];
    const float* v_txt  = (const float*)d_in[5];
    const float* mod_w  = (const float*)d_in[6];
    const float* mod_b  = (const float*)d_in[7];
    const float* qkv_w  = (const float*)d_in[8];
    const float* qkv_b  = (const float*)d_in[9];
    const float* qn_w   = (const float*)d_in[10];
    const float* kn_w   = (const float*)d_in[11];
    const float* proj_w = (const float*)d_in[12];
    const float* proj_b = (const float*)d_in[13];
    const float* mlp_w1 = (const float*)d_in[14];
    const float* mlp_b1 = (const float*)d_in[15];
    const float* mlp_w2 = (const float*)d_in[16];
    const float* mlp_b2 = (const float*)d_in[17];

    char* ws = (char*)d_ws;
    float* sv    = (float*)(ws + OFF_SV);
    float* mod   = (float*)(ws + OFF_MOD);
    bf16*  wqkv  = (bf16*) (ws + OFF_WQKV);
    bf16*  wproj = (bf16*) (ws + OFF_WPROJ);
    bf16*  wmlp1 = (bf16*) (ws + OFF_WMLP1);
    bf16*  wmlp2 = (bf16*) (ws + OFF_WMLP2);
    bf16*  x1    = (bf16*) (ws + OFF_X1);
    float* qkv   = (float*)(ws + OFF_QKV);
    bf16*  attn  = (bf16*) (ws + OFF_ATTN);
    float* proj  = (float*)(ws + OFF_PROJ);
    float* resid = (float*)(ws + OFF_RESID);
    bf16*  x2    = (bf16*) (ws + OFF_X2);
    bf16*  hbuf  = (bf16*) (ws + OFF_H);
    float* mlp2  = (float*)(ws + OFF_MLP2);
    float* out   = (float*)d_out;

    // weight transpose + convert (one-shot, independent of activations)
    k_wt<<<dim3((3 * D_MODEL) / 32, D_MODEL / 32), 256, 0, stream>>>(qkv_w,  wqkv,  D_MODEL, 3 * D_MODEL);
    k_wt<<<dim3(D_MODEL / 32,       D_MODEL / 32), 256, 0, stream>>>(proj_w, wproj, D_MODEL, D_MODEL);
    k_wt<<<dim3(FF_DIM / 32,        D_MODEL / 32), 256, 0, stream>>>(mlp_w1, wmlp1, D_MODEL, FF_DIM);
    k_wt<<<dim3(D_MODEL / 32,       FF_DIM / 32),  256, 0, stream>>>(mlp_w2, wmlp2, FF_DIM,  D_MODEL);

    // modulation vector
    k_silu<<<D_MODEL / 256, 256, 0, stream>>>(vec, sv);
    k_mod<<<(6 * D_MODEL) / 256, 256, 0, stream>>>(sv, mod_w, mod_b, mod);

    // LN + modulation -> x1 bf16
    k_ln_mod<<<S_LEN, 256, 0, stream>>>(img, mod, x1);

    // qkv = x1 @ qkv_w + qkv_b
    k_gemm<0><<<dim3((3 * D_MODEL) / 128, S_LEN / 128), 256, 0, stream>>>(
        x1, wqkv, qkv_b, qkv, S_LEN, 3 * D_MODEL, D_MODEL);

    // RMSNorm + RoPE in place
    k_qknorm_rope<<<(S_LEN * NHEADS) / 8, 256, 0, stream>>>(qkv, qn_w, kn_w, cosb, sinb);

    // WMMA flash attention
    k_attn<<<dim3(NHEADS, S_LEN / 64), 256, 0, stream>>>(qkv, k_txt, v_txt, attn);

    // proj
    k_gemm<0><<<dim3(D_MODEL / 128, S_LEN / 128), 256, 0, stream>>>(
        attn, wproj, proj_b, proj, S_LEN, D_MODEL, D_MODEL);

    // residual + LN
    k_resid_ln<<<S_LEN, 256, 0, stream>>>(img, proj, mod, resid, x2);

    // mlp1 + gelu -> bf16
    k_gemm<1><<<dim3(FF_DIM / 128, S_LEN / 128), 256, 0, stream>>>(
        x2, wmlp1, mlp_b1, hbuf, S_LEN, FF_DIM, D_MODEL);

    // mlp2
    k_gemm<0><<<dim3(D_MODEL / 128, S_LEN / 128), 256, 0, stream>>>(
        hbuf, wmlp2, mlp_b2, mlp2, S_LEN, D_MODEL, FF_DIM);

    // final gated residual
    k_final<<<(S_LEN * D_MODEL) / 256, 256, 0, stream>>>(resid, mlp2, mod, out);
}
